// WassersteinLoss_16028817949388
// MI455X (gfx1250) — matrix operand order repaired and checked by reading, not compiled, not run on hardware
//
#include <hip/hip_runtime.h>

// ---------------- problem constants ----------------
#define N_PTS   8192
#define KDIM    256
#define N_ITER  50
#define JSPLIT  8                       // split of the j (column) dimension per pass
#define ROWS_PER_WG 128                 // 8 waves * 16 rows
#define TILES_PER_CHUNK ((N_PTS / JSPLIT) / 16)   // 64 j-tiles of 16 cols per chunk

#define LOG2E 1.4426950408889634f
#define LN2   0.6931471805599453f
#define LOGB  (-9.010913347279288f)     // log(1/8192) = -log(8192), uniform weights

typedef _Float16 h16;
typedef __attribute__((ext_vector_type(16))) _Float16 v16h;
typedef __attribute__((ext_vector_type(8)))  _Float16 v8h;
typedef __attribute__((ext_vector_type(8)))  float    v8f;

// raw v_exp_f32 / v_log_f32 (base-2); args here are <= 0, underflow->0 is desired
__device__ __forceinline__ float fexp2(float x) { return __builtin_amdgcn_exp2f(x); }
__device__ __forceinline__ float flog2(float x) { return __builtin_amdgcn_logf(x); }

#if __has_builtin(__builtin_amdgcn_sched_group_barrier)
#define SCHED_GROUP(mask, n) __builtin_amdgcn_sched_group_barrier((mask), (n), 0)
#else
#define SCHED_GROUP(mask, n)
#endif

// ---------------------------------------------------------------------------
// f32 -> f16 conversion (optionally scaled) + exact f32 squared row norms.
// One wave32 per row; blockDim 256 -> 8 rows per block.
// ---------------------------------------------------------------------------
__global__ __launch_bounds__(256)
void convert_rows(const float* __restrict__ src, h16* __restrict__ dst,
                  float* __restrict__ norms, float scale) {
    const int wave = threadIdx.x >> 5;
    const int lane = threadIdx.x & 31;
    const int row  = blockIdx.x * 8 + wave;
    const float* r = src + (size_t)row * KDIM;
    h16* d = dst + (size_t)row * KDIM;
    float acc = 0.f;
#pragma unroll
    for (int i = 0; i < KDIM / 32; ++i) {
        float v = r[lane + 32 * i];
        acc += v * v;
        d[lane + 32 * i] = (h16)(v * scale);
    }
#pragma unroll
    for (int k = 16; k >= 1; k >>= 1) acc += __shfl_xor(acc, k, 32);
    if (lane == 0) norms[row] = acc;
}

// c0_j = (0 - ||y_j||^2 + log b) * log2(e)   (g starts at 0)
__global__ void init_cvec(const float* __restrict__ ny, float* __restrict__ c) {
    int i = blockIdx.x * blockDim.x + threadIdx.x;
    c[i] = (LOGB - ny[i]) * LOG2E;
}

// ---------------------------------------------------------------------------
// Fused GEMM + online row-LSE pass (flash-attention style).
// For each row i of Q (vs. all rows j of Kmat in this chunk):
//   u_ij = c'_j + (2*log2e) * q_i . k_j        (scale folded into Q's f16 copy)
//   partials: running (max, sum 2^(u-max)) over j, per chunk.
// Q rows are WMMA A fragments held in registers; K tiles staged via LDS,
// shared by all 8 waves, double-buffered with register prefetch. Scheduling
// groups pin the pipeline: stage stores -> global prefetch -> all 16 B-fragment
// ds reads -> 8 back-to-back WMMAs, so LDS latency hides under the matrix pipe.
// ---------------------------------------------------------------------------
__global__ __launch_bounds__(256)
void lse_pass(const h16* __restrict__ Q, const h16* __restrict__ Kmat,
              const float* __restrict__ cvec,
              float* __restrict__ partM, float* __restrict__ partS) {
    __shared__ __align__(16) h16 smem[2][16][KDIM + 8];   // +8 pad: bank-conflict-free

    const int tid   = threadIdx.x;
    const int wave  = tid >> 5;
    const int lane  = tid & 31;
    const int half  = lane >> 4;           // 0: lanes 0-15, 1: lanes 16-31
    const int l15   = lane & 15;
    const int r0    = blockIdx.x * ROWS_PER_WG + wave * 16;
    const int j0b   = blockIdx.y * (N_PTS / JSPLIT);

    // --- preload A fragments: 16 rows x 256 K for this wave (ISA 16-bit A layout:
    //     lanes<16 hold K {0..7,16..23}+32kc, lanes>=16 hold K {8..15,24..31}+32kc)
    const h16* qrow = Q + (size_t)(r0 + l15) * KDIM;
    const int sel8 = half * 8;
    v16h afrag[8];
#pragma unroll
    for (int kc = 0; kc < 8; ++kc) {
        v8h lo = *(const v8h*)(qrow + kc * 32 + sel8);
        v8h hi = *(const v8h*)(qrow + kc * 32 + 16 + sel8);
        afrag[kc] = __builtin_shufflevector(lo, hi, 0,1,2,3,4,5,6,7,8,9,10,11,12,13,14,15);
    }

    float m[8], s[8];
#pragma unroll
    for (int v = 0; v < 8; ++v) { m[v] = -1e30f; s[v] = 0.f; }

    // --- staging: 256 threads * 32B = one 16x256 f16 tile (8 KB)
    const int strow = tid >> 4;
    const int stcol = (tid & 15) * 16;
    const h16* gsrc0 = Kmat + (size_t)(j0b + strow) * KDIM + stcol;

    uint4 pre0 = *(const uint4*)(gsrc0);
    uint4 pre1 = *(const uint4*)(gsrc0 + 8);

    const int sel16 = half * 16;           // unused: B reads use sel8, same K order as A

    for (int t = 0; t < TILES_PER_CHUNK; ++t) {
        const int buf = t & 1;
        *(uint4*)(&smem[buf][strow][stcol])     = pre0;
        *(uint4*)(&smem[buf][strow][stcol + 8]) = pre1;
        if (t + 1 < TILES_PER_CHUNK) {
            const h16* gsrc = gsrc0 + (size_t)(t + 1) * 16 * KDIM;
            pre0 = *(const uint4*)(gsrc);
            pre1 = *(const uint4*)(gsrc + 8);
        }
        const float cval = cvec[j0b + t * 16 + l15];   // column bias, in base-2 units
        __syncthreads();

        // prefetch all 8 B fragments so ds_loads pipeline ahead of the WMMA chain
        v16h bfrag[8];
#pragma unroll
        for (int kc = 0; kc < 8; ++kc) {
            const h16* bp = &smem[buf][l15][kc * 32 + sel8];
            v8h blo = *(const v8h*)(bp);
            v8h bhi = *(const v8h*)(bp + 16);
            bfrag[kc] = __builtin_shufflevector(blo, bhi, 0,1,2,3,4,5,6,7,8,9,10,11,12,13,14,15);
        }

        // WMMA accumulator pre-seeded with c'_j broadcast down the column
        v8f acc = { cval, cval, cval, cval, cval, cval, cval, cval };
#pragma unroll
        for (int kc = 0; kc < 8; ++kc)
            acc = __builtin_amdgcn_wmma_f32_16x16x32_f16(
                      false, afrag[kc], false, bfrag[kc], (short)0, acc, false, false);

        // online base-2 LSE update (per row, per lane-column stream)
#pragma unroll
        for (int v = 0; v < 8; ++v) {
            float val = acc[v];
            float mn  = fmaxf(m[v], val);
            s[v] = s[v] * fexp2(m[v] - mn) + fexp2(val - mn);
            m[v] = mn;
        }

        // --- pipeline shape for this scheduling region (between the barriers):
        SCHED_GROUP(0x200, 2);    // DS writes: stage next tile into the other buffer
        SCHED_GROUP(0x020, 3);    // VMEM reads: next-tile global prefetch + bias load
        SCHED_GROUP(0x100, 16);   // DS reads: all 16 B-fragment ds_load_b128
        SCHED_GROUP(0x008, 8);    // 8 WMMAs issue back-to-back
    }

    // merge the 16 per-column streams of each row (within each 16-lane half)
#pragma unroll
    for (int k = 1; k < 16; k <<= 1) {
#pragma unroll
        for (int v = 0; v < 8; ++v) {
            float m2 = __shfl_xor(m[v], k, 32);
            float s2 = __shfl_xor(s[v], k, 32);
            float mn = fmaxf(m[v], m2);
            s[v] = s[v] * fexp2(m[v] - mn) + s2 * fexp2(m2 - mn);
            m[v] = mn;
        }
    }
    if (l15 == 0) {
#pragma unroll
        for (int v = 0; v < 8; ++v) {
            int row = r0 + half * 8 + v;     // C layout: lanes>=16 hold rows M=8..15
            partM[row * JSPLIT + blockIdx.y] = m[v];
            partS[row * JSPLIT + blockIdx.y] = s[v];
        }
    }
}

// ---------------------------------------------------------------------------
// Merge the JSPLIT chunk partials:  L2 = log2 sum_j 2^u
//   outF_i   = ||.||^2 - ln2 * L2            (the new f or g)
//   outVec_i = log(w)*log2e - L2             (the next pass's c'/d' vector)
// ---------------------------------------------------------------------------
__global__ void lse_combine(const float* __restrict__ partM, const float* __restrict__ partS,
                            const float* __restrict__ norms,
                            float* __restrict__ outF, float* __restrict__ outVec) {
    int i = blockIdx.x * blockDim.x + threadIdx.x;
    float M = -1e30f, S = 0.f;
#pragma unroll
    for (int c = 0; c < JSPLIT; ++c) {
        float mm = partM[i * JSPLIT + c];
        float ss = partS[i * JSPLIT + c];
        float mn = fmaxf(M, mm);
        S = S * fexp2(M - mn) + ss * fexp2(mm - mn);
        M = mn;
    }
    float L2 = M + flog2(S);
    outF[i]   = norms[i] - LN2 * L2;
    outVec[i] = LOGB * LOG2E - L2;
}

// value = (1/n) [ sum_i f_i * e^{f_i - f'_i} + sum_j g_j ];  out = sqrt(value)
__global__ __launch_bounds__(256)
void final_value(const float* __restrict__ f, const float* __restrict__ fp,
                 const float* __restrict__ g, float* __restrict__ out) {
    __shared__ float red[256];
    float acc = 0.f;
    for (int i = threadIdx.x; i < N_PTS; i += 256)
        acc += f[i] * fexp2((f[i] - fp[i]) * LOG2E) + g[i];
    red[threadIdx.x] = acc;
    __syncthreads();
    for (int o = 128; o > 0; o >>= 1) {
        if (threadIdx.x < o) red[threadIdx.x] += red[threadIdx.x + o];
        __syncthreads();
    }
    if (threadIdx.x == 0) out[0] = sqrtf(red[0] / (float)N_PTS);
}

// ---------------------------------------------------------------------------
extern "C" void kernel_launch(void* const* d_in, const int* in_sizes, int n_in,
                              void* d_out, int out_size, void* d_ws, size_t ws_size,
                              hipStream_t stream) {
    const float* X = (const float*)d_in[0];
    const float* Y = (const float*)d_in[1];
    float* out = (float*)d_out;

    char* ws = (char*)d_ws;
    size_t off = 0;
    auto alloc = [&](size_t bytes) { void* p = ws + off; off += (bytes + 255) & ~size_t(255); return p; };
    h16*   Xh    = (h16*)  alloc((size_t)N_PTS * KDIM * sizeof(h16));   // scaled by 2*log2e
    h16*   Yh    = (h16*)  alloc((size_t)N_PTS * KDIM * sizeof(h16));
    float* nx    = (float*)alloc(N_PTS * sizeof(float));
    float* ny    = (float*)alloc(N_PTS * sizeof(float));
    float* cvec  = (float*)alloc(N_PTS * sizeof(float));
    float* dvec  = (float*)alloc(N_PTS * sizeof(float));
    float* fbuf  = (float*)alloc(N_PTS * sizeof(float));
    float* gbuf  = (float*)alloc(N_PTS * sizeof(float));
    float* fpbuf = (float*)alloc(N_PTS * sizeof(float));
    float* partM = (float*)alloc((size_t)N_PTS * JSPLIT * sizeof(float));
    float* partS = (float*)alloc((size_t)N_PTS * JSPLIT * sizeof(float));

    const dim3 lgrid(N_PTS / ROWS_PER_WG, JSPLIT);   // 64 x 8 workgroups
    const int  vblk = N_PTS / 256;

    convert_rows<<<N_PTS / 8, 256, 0, stream>>>(X, Xh, nx, 2.0f * LOG2E);
    convert_rows<<<N_PTS / 8, 256, 0, stream>>>(Y, Yh, ny, 1.0f);
    init_cvec   <<<vblk, 256, 0, stream>>>(ny, cvec);

    for (int it = 0; it < N_ITER; ++it) {
        // f-update: rows = X, columns = Y, bias = c
        lse_pass   <<<lgrid, 256, 0, stream>>>(Xh, Yh, cvec, partM, partS);
        lse_combine<<<vblk, 256, 0, stream>>>(partM, partS, nx, fbuf, dvec);
        // g-update: rows = Y, columns = X, bias = d
        lse_pass   <<<lgrid, 256, 0, stream>>>(Yh, Xh, dvec, partM, partS);
        lse_combine<<<vblk, 256, 0, stream>>>(partM, partS, ny, gbuf, cvec);
    }
    // one extra f half-update -> f' (gives exact row sums of P)
    lse_pass   <<<lgrid, 256, 0, stream>>>(Xh, Yh, cvec, partM, partS);
    lse_combine<<<vblk, 256, 0, stream>>>(partM, partS, nx, fpbuf, dvec);

    final_value<<<1, 256, 0, stream>>>(fbuf, fpbuf, gbuf, out);
}
